// SingleHeadAttention_55448027791665
// MI455X (gfx1250) — matrix operand order repaired and checked by reading, not compiled, not run on hardware
//
#include <hip/hip_runtime.h>

// ---------------------------------------------------------------------------
// Types for CDNA5 WMMA (wave32): 16x16x32 bf16 -> f32
// ---------------------------------------------------------------------------
typedef __attribute__((ext_vector_type(16))) __bf16 v16bf;
typedef __attribute__((ext_vector_type(8)))  __bf16 v8bf;
typedef __attribute__((ext_vector_type(8)))  float  v8f;

__device__ inline v8f wmma_bf16(v16bf a, v16bf b, v8f c) {
    // (neg_a, A, neg_b, B, c_mod, C, reuse_a, reuse_b)
    return __builtin_amdgcn_wmma_f32_16x16x32_bf16(false, a, false, b,
                                                   (short)0, c, false, false);
}

// A fragment: 16x32 bf16 tile of row-major matrix (ld elements per row).
// Lane (hi = lane>>4, r = lane&15) holds row (row0+r),
// K chunks [k0+hi*8, +8) and [k0+16+hi*8, +8)  (ISA 16-bit A layout).
__device__ inline v16bf load_a(const __bf16* __restrict__ base, int ld,
                               int row0, int k0, int lane) {
    const int hi = (lane >> 4) & 1;
    const int r  = lane & 15;
    const __bf16* p = base + (size_t)(row0 + r) * ld + (k0 + hi * 8);
    v8bf c0 = *(const v8bf*)p;
    v8bf c1 = *(const v8bf*)(p + 16);
    v16bf o;
#pragma unroll
    for (int i = 0; i < 8; ++i) { o[i] = c0[i]; o[8 + i] = c1[i]; }
    return o;
}

// B fragment: 32x16 (KxN) where B[k][n] = Wrow[n][k] with W row-major [N][K].
// Lane (hi, n = lane&15) holds row (n0+n), K chunks [k0+hi*8, +8) and [k0+16+hi*8, +8).
__device__ inline v16bf load_b(const __bf16* __restrict__ base, int ld,
                               int n0, int k0, int lane) {
    const int hi = (lane >> 4) & 1;
    const int n  = lane & 15;
    const __bf16* p = base + (size_t)(n0 + n) * ld + (k0 + hi * 8);
    v8bf c0 = *(const v8bf*)p;
    v8bf c1 = *(const v8bf*)(p + 16);
    v16bf o;
#pragma unroll
    for (int i = 0; i < 8; ++i) { o[i] = c0[i]; o[8 + i] = c1[i]; }
    return o;
}

// ---------------------------------------------------------------------------
// Kernel 1: f32 -> bf16 conversion
// ---------------------------------------------------------------------------
__global__ void cvt_f32_bf16(const float* __restrict__ x, __bf16* __restrict__ y,
                             long long n) {
    long long i = (long long)blockIdx.x * blockDim.x + threadIdx.x;
    if (i < n) y[i] = (__bf16)x[i];
}

// ---------------------------------------------------------------------------
// Kernel 2: generic C = A(MxK) * W(NxK)^T  GEMM on WMMA.
// Block = 128 threads = 4 waves (2x2); wave tile 32x64; block tile 64x128.
// MODE 0: bf16 output row-major (ldc = row stride)
// MODE 1: bf16 output TRANSPOSED (C[n][m], ldc = M-dimension stride)
// MODE 2: f32 output row-major
// ---------------------------------------------------------------------------
template <int MODE>
__global__ void gemm_wmma_kernel(const __bf16* __restrict__ A,
                                 const __bf16* __restrict__ W,
                                 void* __restrict__ C, int K, int ldc) {
    const int lane = threadIdx.x & 31;
    const int wave = threadIdx.x >> 5;
    const int wm = wave & 1, wn = wave >> 1;
    const int m0 = blockIdx.y * 64 + wm * 32;
    const int n0 = blockIdx.x * 128 + wn * 64;

    v8f acc[2][4] = {};
    for (int k0 = 0; k0 < K; k0 += 32) {
        v16bf a0 = load_a(A, K, m0, k0, lane);
        v16bf a1 = load_a(A, K, m0 + 16, k0, lane);
        v16bf bb[4];
#pragma unroll
        for (int j = 0; j < 4; ++j) bb[j] = load_b(W, K, n0 + 16 * j, k0, lane);
#pragma unroll
        for (int j = 0; j < 4; ++j) {
            acc[0][j] = wmma_bf16(a0, bb[j], acc[0][j]);
            acc[1][j] = wmma_bf16(a1, bb[j], acc[1][j]);
        }
    }

    const int col = lane & 15, hi = (lane >> 4) & 1;
#pragma unroll
    for (int i = 0; i < 2; ++i)
#pragma unroll
        for (int j = 0; j < 4; ++j)
#pragma unroll
            for (int r = 0; r < 8; ++r) {
                const int row = m0 + 16 * i + 8 * hi + r;
                const int cc  = n0 + 16 * j + col;
                if constexpr (MODE == 0) {
                    ((__bf16*)C)[(size_t)row * ldc + cc] = (__bf16)acc[i][j][r];
                } else if constexpr (MODE == 1) {
                    ((__bf16*)C)[(size_t)cc * ldc + row] = (__bf16)acc[i][j][r];
                } else {
                    ((float*)C)[(size_t)row * ldc + cc] = acc[i][j][r];
                }
            }
}

// ---------------------------------------------------------------------------
// Kernel 3: causal masked scaled scores.
// One wave per 16-query x 64-key strip; A fragment reused across 4 key tiles
// (4 WMMAs per 5 fragment loads). Strips fully above the diagonal exit early;
// masked entries get -inf (softmax never reads past the causal boundary).
// Sbuf[b][q][k] = Q.K^T / sqrt(D)
// ---------------------------------------------------------------------------
__global__ void scores_kernel(const __bf16* __restrict__ Q,
                              const __bf16* __restrict__ Kmat,
                              float* __restrict__ Sbuf, int S, int D) {
    const int jx = blockIdx.x;          // 64-key strip index
    const int qt = blockIdx.y;          // 16-query tile index
    const int b  = blockIdx.z;
    const int j0 = jx * 64;
    const int q0 = qt * 16;
    if (j0 > q0 + 15) return;           // strip entirely above diagonal

    const int lane = threadIdx.x & 31;
    const __bf16* Qb = Q    + (size_t)b * S * D;
    const __bf16* Kb = Kmat + (size_t)b * S * D;

    v8f acc[4] = {};
    for (int k0 = 0; k0 < D; k0 += 32) {
        v16bf a = load_a(Qb, D, q0, k0, lane);
#pragma unroll
        for (int j = 0; j < 4; ++j) {
            v16bf bb = load_b(Kb, D, j0 + 16 * j, k0, lane);
            acc[j] = wmma_bf16(a, bb, acc[j]);
        }
    }

    const float scale = rsqrtf((float)D);
    const int col = lane & 15, hi = (lane >> 4) & 1;
#pragma unroll
    for (int j = 0; j < 4; ++j)
#pragma unroll
        for (int r = 0; r < 8; ++r) {
            const int q = q0 + 8 * hi + r;
            const int k = j0 + 16 * j + col;
            float s = acc[j][r] * scale;
            if (k > q) s = -__builtin_inff();
            Sbuf[((size_t)b * S + q) * S + k] = s;
        }
}

// ---------------------------------------------------------------------------
// Kernel 4: row softmax -> bf16 probabilities P, zero-padded to 32-col step.
// One 256-thread block per (b, q) row.
// ---------------------------------------------------------------------------
__global__ void softmax_kernel(const float* __restrict__ Sbuf,
                               __bf16* __restrict__ P, int S) {
    __shared__ float red[256];
    const int tid = threadIdx.x;
    const int gs  = blockIdx.x;          // b*S + s
    const int s   = gs % S;
    const float* row = Sbuf + (size_t)gs * S;
    __bf16* prow     = P    + (size_t)gs * S;
    const int kc = s + 1;                // valid (causal) columns

    float m = -__builtin_inff();
    for (int c = tid; c < kc; c += 256) m = fmaxf(m, row[c]);
    red[tid] = m; __syncthreads();
    for (int off = 128; off > 0; off >>= 1) {
        if (tid < off) red[tid] = fmaxf(red[tid], red[tid + off]);
        __syncthreads();
    }
    m = red[0]; __syncthreads();

    float l = 0.0f;
    for (int c = tid; c < kc; c += 256) l += __expf(row[c] - m);
    red[tid] = l; __syncthreads();
    for (int off = 128; off > 0; off >>= 1) {
        if (tid < off) red[tid] += red[tid + off];
        __syncthreads();
    }
    const float invl = 1.0f / red[0];

    const int qt = s >> 4;
    const int K2 = ((qt + 2) & ~1) * 16;  // tile K-range rounded up to 32
    for (int c = tid; c < K2; c += 256) {
        float pv = (c < kc) ? __expf(row[c] - m) * invl : 0.0f;
        prow[c] = (__bf16)pv;
    }
}

// ---------------------------------------------------------------------------
// Kernel 5: context = P @ V, using Vt[d][b*S+s] (transposed V) as B-operand.
// Wave tile 16x64 (M-tile fixed to one query tile so K-limit is uniform).
// kSteps = ceil((qt+1)*16 / 32) = (qt+2)>>1 exploits causality.
// ---------------------------------------------------------------------------
__global__ void ctx_kernel(const __bf16* __restrict__ P,
                           const __bf16* __restrict__ Vt,
                           __bf16* __restrict__ Ctx, int S, int D, int Mtot) {
    const int lane = threadIdx.x & 31;
    const int wave = threadIdx.x >> 5;
    const int qtiles = S >> 4;
    const int b   = blockIdx.y / qtiles;
    const int qtl = blockIdx.y % qtiles;
    const int row0 = b * S + qtl * 16;   // global row into P / Ctx
    const int n0 = blockIdx.x * 256 + wave * 64;
    const int kSteps = (qtl + 2) >> 1;

    v8f acc[4] = {};
    for (int kk = 0; kk < kSteps; ++kk) {
        const int k0 = kk * 32;
        v16bf a = load_a(P, S, row0, k0, lane);
#pragma unroll
        for (int j = 0; j < 4; ++j) {
            v16bf bb = load_b(Vt, Mtot, n0 + 16 * j, b * S + k0, lane);
            acc[j] = wmma_bf16(a, bb, acc[j]);
        }
    }

    const int col = lane & 15, hi = (lane >> 4) & 1;
#pragma unroll
    for (int j = 0; j < 4; ++j)
#pragma unroll
        for (int r = 0; r < 8; ++r)
            Ctx[(size_t)(row0 + 8 * hi + r) * D + n0 + 16 * j + col] =
                (__bf16)acc[j][r];
}

// ---------------------------------------------------------------------------
// Launch
// ---------------------------------------------------------------------------
extern "C" void kernel_launch(void* const* d_in, const int* in_sizes, int n_in,
                              void* d_out, int out_size, void* d_ws, size_t ws_size,
                              hipStream_t stream) {
    (void)in_sizes; (void)n_in; (void)out_size; (void)ws_size;
    constexpr int B = 4, S = 2048, D = 1024;
    constexpr int M = B * S;  // 8192 rows total

    const float* z_f  = (const float*)d_in[0];
    const float* wq_f = (const float*)d_in[1];
    const float* wk_f = (const float*)d_in[2];
    const float* wv_f = (const float*)d_in[3];
    const float* wo_f = (const float*)d_in[4];

    // ---- workspace carve (256B aligned) ----
    size_t off = 0;
    auto carve = [&](size_t bytes) -> void* {
        void* p = (char*)d_ws + off;
        off += (bytes + 255) & ~(size_t)255;
        return p;
    };
    __bf16* zb   = (__bf16*)carve((size_t)M * D * 2);   // 16 MB
    __bf16* Qb   = (__bf16*)carve((size_t)M * D * 2);   // 16 MB
    __bf16* Kb   = (__bf16*)carve((size_t)M * D * 2);   // 16 MB (contig after Qb)
    __bf16* Vt   = (__bf16*)carve((size_t)D * M * 2);   // 16 MB, [D][B*S]
    __bf16* wqb  = (__bf16*)carve((size_t)D * D * 2);
    __bf16* wkb  = (__bf16*)carve((size_t)D * D * 2);
    __bf16* wvb  = (__bf16*)carve((size_t)D * D * 2);
    __bf16* wob  = (__bf16*)carve((size_t)D * D * 2);
    float*  Sbuf = (float*)carve((size_t)M * S * 4);    // 64 MB
    // Aliases over dead regions:
    __bf16* P   = Qb;              // needs 32 MB  = Qb+Kb (dead after scores)
    __bf16* Ctx = (__bf16*)Sbuf;   // needs 16 MB  < Sbuf (dead after softmax)

    // ---- 1) convert inputs to bf16 ----
    {
        long long nz = (long long)M * D;
        cvt_f32_bf16<<<(int)((nz + 255) / 256), 256, 0, stream>>>(z_f, zb, nz);
        long long nw = (long long)D * D;
        int gw = (int)((nw + 255) / 256);
        cvt_f32_bf16<<<gw, 256, 0, stream>>>(wq_f, wqb, nw);
        cvt_f32_bf16<<<gw, 256, 0, stream>>>(wk_f, wkb, nw);
        cvt_f32_bf16<<<gw, 256, 0, stream>>>(wv_f, wvb, nw);
        cvt_f32_bf16<<<gw, 256, 0, stream>>>(wo_f, wob, nw);
    }

    // ---- 2) QKV projections: X @ W^T ----
    dim3 ggrid(D / 128, M / 64);
    gemm_wmma_kernel<0><<<ggrid, 128, 0, stream>>>(zb, wqb, (void*)Qb, D, D);
    gemm_wmma_kernel<0><<<ggrid, 128, 0, stream>>>(zb, wkb, (void*)Kb, D, D);
    gemm_wmma_kernel<1><<<ggrid, 128, 0, stream>>>(zb, wvb, (void*)Vt, D, M);

    // ---- 3) causal scores (lower-triangular 16x64 strips) ----
    scores_kernel<<<dim3(S / 64, S / 16, B), 32, 0, stream>>>(Qb, Kb, Sbuf, S, D);

    // ---- 4) row softmax -> P (bf16, padded to 32-col multiple) ----
    softmax_kernel<<<M, 256, 0, stream>>>(Sbuf, P, S);

    // ---- 5) context = P @ V (causality-limited K loop) ----
    ctx_kernel<<<dim3(D / 256, M / 16), 128, 0, stream>>>(P, Vt, Ctx, S, D, M);

    // ---- 6) output projection -> f32 d_out ----
    gemm_wmma_kernel<2><<<ggrid, 128, 0, stream>>>(Ctx, wob, d_out, D, D);
}
